// MultiHeadAttentionClassifier_87196426043407
// MI455X (gfx1250) — matrix operand order repaired and checked by reading, not compiled, not run on hardware
//
#include <hip/hip_runtime.h>
#include <hip/hip_bf16.h>

// ---------------- types for CDNA5 WMMA ----------------
typedef __bf16  v16bf __attribute__((ext_vector_type(16)));
typedef __bf16  v8bf  __attribute__((ext_vector_type(8)));
typedef float   v8f   __attribute__((ext_vector_type(8)));

#define D_MODEL 1024
#define N_HEADS 8
#define HEAD_DIM 128
#define T_TOK   8193          // 1 class token + 8192
#define T_PAD   8208          // 513 * 16
#define SCALE   0.08838834764831845f   // 1/sqrt(128)

// ---------------- helpers ----------------
__device__ __forceinline__ unsigned short f2bf(float f) {
    unsigned int u = __float_as_uint(f);
    unsigned int r = u + 0x7FFFu + ((u >> 16) & 1u);   // round-to-nearest-even
    return (unsigned short)(r >> 16);
}

// Build zero-padded bf16 token matrix [T_PAD x 1024]; row0 = class token.
__global__ void build_tokens_bf16(const float* __restrict__ x,
                                  const float* __restrict__ cls,
                                  unsigned short* __restrict__ tok) {
    int idx = blockIdx.x * blockDim.x + threadIdx.x;
    if (idx >= T_PAD * D_MODEL) return;
    int row = idx >> 10, col = idx & 1023;
    float v = 0.f;
    if (row == 0)          v = cls[col];
    else if (row <= 8192)  v = x[(size_t)(row - 1) * D_MODEL + col];
    tok[idx] = f2bf(v);
}

__global__ void f32_to_bf16(const float* __restrict__ src,
                            unsigned short* __restrict__ dst, int n) {
    int idx = blockIdx.x * blockDim.x + threadIdx.x;
    if (idx < n) dst[idx] = f2bf(src[idx]);
}

// ---------------- bf16 WMMA GEMM:  C[M x 1024] = A[M x 1024] @ W^T + bias ----
// A is bf16 row-major (pitch 1024). W is bf16 row-major [1024 x 1024] so that
// B = W^T is column-major -> per-lane B fragment loads are contiguous in K.
// One wave computes a 16x64 output strip (4 WMMA tiles, A fragment reused 4x).
__global__ __launch_bounds__(128) void gemm_bf16_wmma(
    const __bf16* __restrict__ A, const __bf16* __restrict__ W,
    const float* __restrict__ bias, float* __restrict__ C, int Mvalid) {
    const int lane  = threadIdx.x & 31;
    const int wave  = threadIdx.x >> 5;
    const int half  = lane >> 4;        // K-half selector
    const int l15   = lane & 15;        // M row (A) / N col (B,C)
    const int tm    = blockIdx.y;                       // 0..512
    const int tnb   = (blockIdx.x * 4 + wave) * 4;      // first of 4 n-tiles

    const __bf16* Arow  = A + (size_t)(tm * 16 + l15) * D_MODEL;
    const __bf16* Brow0 = W + (size_t)(tnb * 16 + l15) * D_MODEL;

    v8f c[4] = {};
    for (int k0 = 0; k0 < D_MODEL; k0 += 32) {
        __builtin_prefetch(Arow + k0 + 256, 0, 0);      // global_prefetch_b8
        // A fragment (16x32): lane l15=M, half selects K sub-block of 8
        v16bf a;
        {
            v8bf a0 = *(const v8bf*)(Arow + k0 + half * 8);
            v8bf a1 = *(const v8bf*)(Arow + k0 + 16 + half * 8);
            #pragma unroll
            for (int i = 0; i < 8; ++i) { a[i] = a0[i]; a[8 + i] = a1[i]; }
        }
        #pragma unroll
        for (int j = 0; j < 4; ++j) {
            const __bf16* Brow = Brow0 + (size_t)j * 16 * D_MODEL;
            __builtin_prefetch(Brow + k0 + 256, 0, 0);
            // B fragment (32x16): lane l15=N, half selects K sub-block of 8
            v8bf b0 = *(const v8bf*)(Brow + k0 + half * 8);
            v8bf b1 = *(const v8bf*)(Brow + k0 + 16 + half * 8);
            v16bf b;
            #pragma unroll
            for (int i = 0; i < 8; ++i) { b[i] = b0[i]; b[8 + i] = b1[i]; }
            c[j] = __builtin_amdgcn_wmma_f32_16x16x32_bf16(
                       false, a, false, b, (short)0, c[j], false, false);
        }
    }
    // Store: VGPR r -> M = half*8 + r, N = l15 (per 16x16 f32 C/D layout)
    #pragma unroll
    for (int j = 0; j < 4; ++j) {
        int n = (tnb + j) * 16 + l15;
        float bn = bias[n];
        #pragma unroll
        for (int r = 0; r < 8; ++r) {
            int m = tm * 16 + half * 8 + r;
            if (m < Mvalid) C[(size_t)m * D_MODEL + n] = c[j][r] + bn;
        }
    }
}

// out[j] = dot(v, W[j,:]) + b[j], j in [0,1024)
__global__ void vec_matT(const float* __restrict__ v, const float* __restrict__ W,
                         const float* __restrict__ b, float* __restrict__ out) {
    int j = blockIdx.x * blockDim.x + threadIdx.x;
    const float* Wr = W + (size_t)j * D_MODEL;
    float acc = 0.f;
    for (int i = 0; i < D_MODEL; ++i) acc += v[i] * Wr[i];
    out[j] = acc + b[j];
}

// scores[h*T + t] = SCALE * dot(q0_h, K[t, h*128 : h*128+128])
__global__ void scores_kernel(const float* __restrict__ q0,
                              const float* __restrict__ K,
                              float* __restrict__ S) {
    int idx = blockIdx.x * blockDim.x + threadIdx.x;
    if (idx >= N_HEADS * T_TOK) return;
    int h = idx / T_TOK, t = idx - h * T_TOK;
    const float* q = q0 + h * HEAD_DIM;
    const float* k = K + (size_t)t * D_MODEL + h * HEAD_DIM;
    float acc = 0.f;
    for (int i = 0; i < HEAD_DIM; ++i) acc += q[i] * k[i];
    S[idx] = acc * SCALE;
}

// per-head max and sum-of-exp over T_TOK
__global__ __launch_bounds__(256) void softmax_stats(const float* __restrict__ S,
                                                     float* __restrict__ stats) {
    __shared__ float red[256];
    int h = blockIdx.x, tid = threadIdx.x;
    const float* s = S + (size_t)h * T_TOK;
    float mx = -3.4e38f;
    for (int t = tid; t < T_TOK; t += 256) mx = fmaxf(mx, s[t]);
    red[tid] = mx; __syncthreads();
    for (int w = 128; w > 0; w >>= 1) {
        if (tid < w) red[tid] = fmaxf(red[tid], red[tid + w]);
        __syncthreads();
    }
    mx = red[0]; __syncthreads();
    float sum = 0.f;
    for (int t = tid; t < T_TOK; t += 256) sum += __expf(s[t] - mx);
    red[tid] = sum; __syncthreads();
    for (int w = 128; w > 0; w >>= 1) {
        if (tid < w) red[tid] += red[tid + w];
        __syncthreads();
    }
    if (tid == 0) { stats[2 * h] = mx; stats[2 * h + 1] = red[0]; }
}

// attn0[h*128+d] = sum_t softmax(scores_h)[t] * V[t, h*128+d]
__global__ __launch_bounds__(256) void attn_out_kernel(const float* __restrict__ S,
                                                       const float* __restrict__ stats,
                                                       const float* __restrict__ V,
                                                       float* __restrict__ attn0) {
    __shared__ float red[256];
    int hd = blockIdx.x, h = hd >> 7, d = hd & 127;
    float mx = stats[2 * h];
    const float* s = S + (size_t)h * T_TOK;
    const float* v = V + h * HEAD_DIM + d;
    float acc = 0.f;
    for (int t = threadIdx.x; t < T_TOK; t += 256)
        acc += __expf(s[t] - mx) * v[(size_t)t * D_MODEL];
    red[threadIdx.x] = acc; __syncthreads();
    for (int w = 128; w > 0; w >>= 1) {
        if (threadIdx.x < w) red[threadIdx.x] += red[threadIdx.x + w];
        __syncthreads();
    }
    if (threadIdx.x == 0) attn0[hd] = red[0] / stats[2 * h + 1];
}

// A[t] = (1/8) * sum_h softmax(scores_h)[t+1]
__global__ void attnA_kernel(const float* __restrict__ S,
                             const float* __restrict__ stats,
                             float* __restrict__ Aout) {
    int t = blockIdx.x * blockDim.x + threadIdx.x;
    if (t >= 8192) return;
    float acc = 0.f;
    #pragma unroll
    for (int h = 0; h < N_HEADS; ++h)
        acc += __expf(S[(size_t)h * T_TOK + t + 1] - stats[2 * h]) / stats[2 * h + 1];
    Aout[t] = acc * 0.125f;
}

// logits -> softmax -> outputs [l0,l1, p0,p1, yhat, p0,p1]
__global__ __launch_bounds__(256) void classifier_head(const float* __restrict__ out0,
                                                       const float* __restrict__ Wc,
                                                       const float* __restrict__ bc,
                                                       float* __restrict__ dout) {
    __shared__ float r0[256], r1[256];
    int tid = threadIdx.x;
    float a0 = 0.f, a1 = 0.f;
    for (int i = tid; i < D_MODEL; i += 256) {
        float v = out0[i];
        a0 += v * Wc[i];
        a1 += v * Wc[D_MODEL + i];
    }
    r0[tid] = a0; r1[tid] = a1; __syncthreads();
    for (int w = 128; w > 0; w >>= 1) {
        if (tid < w) { r0[tid] += r0[tid + w]; r1[tid] += r1[tid + w]; }
        __syncthreads();
    }
    if (tid == 0) {
        float l0 = r0[0] + bc[0], l1 = r1[0] + bc[1];
        float m = fmaxf(l0, l1);
        float e0 = __expf(l0 - m), e1 = __expf(l1 - m);
        float s = e0 + e1, p0 = e0 / s, p1 = e1 / s;
        int yhat = (p1 > p0) ? 1 : 0;
        dout[0] = l0; dout[1] = l1;          // top_instance
        dout[2] = p0; dout[3] = p1;          // Y_prob
        dout[4] = (float)yhat;               // Y_hat
        dout[5] = p0; dout[6] = p1;          // y_probs
    }
}

extern "C" void kernel_launch(void* const* d_in, const int* in_sizes, int n_in,
                              void* d_out, int out_size, void* d_ws, size_t ws_size,
                              hipStream_t stream) {
    const float* x   = (const float*)d_in[0];
    const float* cls = (const float*)d_in[1];
    const float* Wq  = (const float*)d_in[2];
    const float* bq  = (const float*)d_in[3];
    const float* Wk  = (const float*)d_in[4];
    const float* bk  = (const float*)d_in[5];
    const float* Wv  = (const float*)d_in[6];
    const float* bv  = (const float*)d_in[7];
    const float* Wo  = (const float*)d_in[8];
    const float* bo  = (const float*)d_in[9];
    const float* Wc  = (const float*)d_in[10];
    const float* bc  = (const float*)d_in[11];
    float* out = (float*)d_out;

    // ---- workspace layout (256B aligned) ----
    char* w = (char*)d_ws;
    constexpr size_t OFF_TOK  = 0;                                   // T_PAD*1024*2
    constexpr size_t OFF_WK   = OFF_TOK + (size_t)T_PAD * 1024 * 2;  // 16,809,984
    constexpr size_t OFF_WV   = OFF_WK + 1024 * 1024 * 2;
    constexpr size_t OFF_K    = OFF_WV + 1024 * 1024 * 2;
    constexpr size_t OFF_V    = OFF_K + (size_t)T_TOK * 1024 * 4;
    constexpr size_t OFF_S    = OFF_V + (size_t)T_TOK * 1024 * 4;
    constexpr size_t OFF_Q0   = ((OFF_S + (size_t)N_HEADS * T_TOK * 4 + 255) & ~(size_t)255);
    constexpr size_t OFF_AT0  = OFF_Q0 + 4096;
    constexpr size_t OFF_OUT0 = OFF_AT0 + 4096;
    constexpr size_t OFF_ST   = OFF_OUT0 + 4096;

    unsigned short* tokU = (unsigned short*)(w + OFF_TOK);
    unsigned short* wkU  = (unsigned short*)(w + OFF_WK);
    unsigned short* wvU  = (unsigned short*)(w + OFF_WV);
    const __bf16*   tokB = (const __bf16*)(w + OFF_TOK);
    const __bf16*   wkB  = (const __bf16*)(w + OFF_WK);
    const __bf16*   wvB  = (const __bf16*)(w + OFF_WV);
    float* Kbuf  = (float*)(w + OFF_K);
    float* Vbuf  = (float*)(w + OFF_V);
    float* Sbuf  = (float*)(w + OFF_S);
    float* q0    = (float*)(w + OFF_Q0);
    float* attn0 = (float*)(w + OFF_AT0);
    float* out0  = (float*)(w + OFF_OUT0);
    float* stats = (float*)(w + OFF_ST);

    // 1) bf16 conversions
    build_tokens_bf16<<<(T_PAD * 1024 + 255) / 256, 256, 0, stream>>>(x, cls, tokU);
    f32_to_bf16<<<(1024 * 1024 + 255) / 256, 256, 0, stream>>>(Wk, wkU, 1024 * 1024);
    f32_to_bf16<<<(1024 * 1024 + 255) / 256, 256, 0, stream>>>(Wv, wvU, 1024 * 1024);

    // 2) q0 for the class token only
    vec_matT<<<4, 256, 0, stream>>>(cls, Wq, bq, q0);

    // 3) K and V projections — the WMMA-dominant GEMMs
    dim3 ggrid(4, T_PAD / 16);   // 4 wave-groups of n-tiles x 513 m-tiles
    gemm_bf16_wmma<<<ggrid, 128, 0, stream>>>(tokB, wkB, bk, Kbuf, T_TOK);
    gemm_bf16_wmma<<<ggrid, 128, 0, stream>>>(tokB, wvB, bv, Vbuf, T_TOK);

    // 4) single-query attention
    scores_kernel<<<(N_HEADS * T_TOK + 255) / 256, 256, 0, stream>>>(q0, Kbuf, Sbuf);
    softmax_stats<<<N_HEADS, 256, 0, stream>>>(Sbuf, stats);
    attn_out_kernel<<<D_MODEL, 256, 0, stream>>>(Sbuf, stats, Vbuf, attn0);
    attnA_kernel<<<8192 / 256, 256, 0, stream>>>(Sbuf, stats, out + 7);

    // 5) output projection (row 0 only) + classifier head
    vec_matT<<<4, 256, 0, stream>>>(attn0, Wo, bo, out0);
    classifier_head<<<1, 256, 0, stream>>>(out0, Wc, bc, out);
}
